// Seq2Seq_40037685134117
// MI455X (gfx1250) — matrix operand order repaired and checked by reading, not compiled, not run on hardware
//
#include <hip/hip_runtime.h>
#include <hip/hip_bf16.h>

// ---------------------------------------------------------------------------
// Seq2Seq GRU + attention for MI455X (gfx1250, wave32, WMMA).
// All matmuls run on v_wmma_f32_16x16x32_f16 (f16 in, f32 accumulate).
// Weights converted once per call to f16 in workspace; fc_W f16 (65MB) stays
// resident in the 192MB L2 across the 127 decoder steps.
// ---------------------------------------------------------------------------

typedef _Float16 half_t;
typedef __attribute__((ext_vector_type(16))) _Float16 v16h;
typedef __attribute__((ext_vector_type(8)))  _Float16 v8h;
typedef __attribute__((ext_vector_type(8)))  float    v8f;

static constexpr int H    = 1024;
static constexpr int H3   = 3072;
static constexpr int Bsz  = 16;
static constexpr int S    = 128;
static constexpr int T    = 128;
static constexpr int V    = 32000;
static constexpr int TNW  = 4;     // 16x16 N-tiles per wave

// ---------------------------------------------------------------------------
// Generic WMMA GEMM: C[M,N] (f32) = A16[M,K] @ W16[N,K]^T + bias[N]
// M is a multiple of 16 (grid.y = M/16); N a multiple of 64.
// Each wave owns TNW adjacent 16x16 output tiles, chained over K.
// ---------------------------------------------------------------------------
__global__ void gemm_wmma_f16(const half_t* __restrict__ A, int lda,
                              const half_t* __restrict__ W, int ldw,
                              const float* __restrict__ bias,
                              float* __restrict__ C, long long ldc,
                              long long coff, int K, int Ntiles)
{
    const int lane   = threadIdx.x & 31;
    const int wave   = threadIdx.x >> 5;
    const int wpb    = blockDim.x >> 5;
    const int ngroup = blockIdx.x * wpb + wave;
    if (ngroup * TNW >= Ntiles) return;            // wave-uniform guard
    const int n0base = ngroup * (TNW * 16);
    const int m0     = blockIdx.y * 16;

    const int row = lane & 15;                     // A row / B col / C col
    const int hs  = lane >> 4;                     // half-select

    v8f acc[TNW];
#pragma unroll
    for (int t = 0; t < TNW; ++t)
        acc[t] = (v8f){0.f,0.f,0.f,0.f,0.f,0.f,0.f,0.f};

    const half_t* Ar = A + (size_t)(m0 + row) * lda + hs * 16;
    const half_t* Wr = W + (size_t)(n0base + row) * ldw + hs * 16;

    for (int k0 = 0; k0 < K; k0 += 32) {
        // A fragment: halves 0..15 = K[k0+hs*16 .. +15] (contiguous)
        v8h alo = *(const v8h*)(Ar + k0);
        v8h ahi = *(const v8h*)(Ar + k0 + 8);
        v16h a;
#pragma unroll
        for (int i = 0; i < 8; ++i) { a[i] = alo[i]; a[8 + i] = ahi[i]; }
#pragma unroll
        for (int t = 0; t < TNW; ++t) {
            // B fragment: lane n holds W[n0+n][k0+hs*16 .. +15] (contiguous 32B)
            v16h b = *(const v16h*)(Wr + (size_t)t * 16 * ldw + k0);
            acc[t] = __builtin_amdgcn_wmma_f32_16x16x32_f16(
                false, a, false, b, (short)0, acc[t], false, false);
        }
    }

#pragma unroll
    for (int t = 0; t < TNW; ++t) {
        const int n0  = n0base + t * 16;
        const int col = lane & 15;
        const float bv = bias ? bias[n0 + col] : 0.f;
#pragma unroll
        for (int j = 0; j < 8; ++j) {
            const int r = j + hs * 8;              // C layout: VGPR j -> M = j (+8)
            C[(size_t)(m0 + r) * ldc + coff + n0 + col] = acc[t][j] + bv;
        }
    }
}

// ---------------------------------------------------------------------------
// Elementwise / small kernels
// ---------------------------------------------------------------------------
__global__ void f32_to_f16_kernel(const float* __restrict__ in,
                                  half_t* __restrict__ out, int n)
{
    int i = blockIdx.x * blockDim.x + threadIdx.x;
    if (i < n) out[i] = (half_t)in[i];
}

__global__ void gather_src_kernel(const int* __restrict__ tok,
                                  const float* __restrict__ emb,
                                  half_t* __restrict__ out16)
{
    int i = blockIdx.x * blockDim.x + threadIdx.x;      // over B*S*H
    if (i >= Bsz * S * H) return;
    int t = i >> 10, c = i & 1023;
    out16[i] = (half_t)emb[(size_t)tok[t] * H + c];
}

__global__ void gather_dec_kernel(const int* __restrict__ tgt,
                                  const float* __restrict__ emb,
                                  float* __restrict__ outF,
                                  half_t* __restrict__ out16)
{
    int i = blockIdx.x * blockDim.x + threadIdx.x;      // over B*127*H
    if (i >= Bsz * (T - 1) * H) return;
    int bt = i >> 10, c = i & 1023;
    int b = bt / (T - 1), t = bt % (T - 1);
    float v = emb[(size_t)tgt[b * T + t] * H + c];
    outF[i] = v;
    out16[i] = (half_t)v;
}

__global__ void init_h_kernel(float* __restrict__ h, half_t* __restrict__ h16)
{
    int i = blockIdx.x * blockDim.x + threadIdx.x;
    if (i < Bsz * H) { h[i] = 0.f; h16[i] = (half_t)0.f; }
}

__global__ void zero_row0_kernel(float* __restrict__ out)
{
    int i = blockIdx.x * blockDim.x + threadIdx.x;      // over B*V
    if (i >= Bsz * V) return;
    int b = i / V, v = i % V;
    out[(size_t)b * T * V + v] = 0.f;
}

__global__ void gru_elem_kernel(const float* __restrict__ gi, int ldgi,
                                const float* __restrict__ gh,
                                float* __restrict__ h, half_t* __restrict__ h16,
                                float* __restrict__ enc_slice, int ldenc)
{
    int i = blockIdx.x * blockDim.x + threadIdx.x;
    if (i >= Bsz * H) return;
    int b = i >> 10, c = i & 1023;
    const float* gib = gi + (size_t)b * ldgi;
    const float* ghb = gh + (size_t)b * H3;
    float r = 1.f / (1.f + __expf(-(gib[c]            + ghb[c])));
    float z = 1.f / (1.f + __expf(-(gib[H + c]        + ghb[H + c])));
    float n = tanhf(gib[2 * H + c] + r * ghb[2 * H + c]);
    float hn = (1.f - z) * n + z * h[i];
    h[i]   = hn;
    h16[i] = (half_t)hn;
    if (enc_slice) enc_slice[(size_t)b * ldenc + c] = hn;
}

// One wave per (b,s): 1024-length dot, wave32 xor-shuffle reduction.
__global__ void attn_scores_kernel(const float* __restrict__ encOut,
                                   const float* __restrict__ qbase, int ldq,
                                   float* __restrict__ scores)
{
    int w    = blockIdx.x * (blockDim.x >> 5) + (threadIdx.x >> 5);
    int lane = threadIdx.x & 31;
    if (w >= Bsz * S) return;
    int b = w >> 7, s = w & 127;
    const float* e = encOut + ((size_t)b * S + s) * H;
    const float* q = qbase + (size_t)b * ldq;
    float sum = 0.f;
    for (int k = lane; k < H; k += 32) sum += e[k] * q[k];
#pragma unroll
    for (int m = 16; m > 0; m >>= 1) sum += __shfl_xor(sum, m, 32);
    if (lane == 0) scores[w] = sum;
}

__global__ void softmax128_kernel(float* __restrict__ scores)
{
    __shared__ float red[128];
    int b = blockIdx.x, t = threadIdx.x;
    float v = scores[b * S + t];
    red[t] = v; __syncthreads();
    for (int off = 64; off > 0; off >>= 1) {
        if (t < off) red[t] = fmaxf(red[t], red[t + off]);
        __syncthreads();
    }
    float mx = red[0]; __syncthreads();
    float e = __expf(v - mx);
    red[t] = e; __syncthreads();
    for (int off = 64; off > 0; off >>= 1) {
        if (t < off) red[t] += red[t + off];
        __syncthreads();
    }
    scores[b * S + t] = e / red[0];
}

// context[b][c] = sum_s attn[b][s] * encOut[b][s][c];  x = [tok_emb, context] (f16)
__global__ void attn_ctx_concat_kernel(const float* __restrict__ attn,
                                       const float* __restrict__ encOut,
                                       const half_t* __restrict__ tok16, int ldt,
                                       half_t* __restrict__ x16)
{
    int i = blockIdx.x * blockDim.x + threadIdx.x;
    if (i >= Bsz * H) return;
    int b = i >> 10, c = i & 1023;
    const float* a = attn + b * S;
    const float* e = encOut + (size_t)b * S * H + c;
    float acc = 0.f;
#pragma unroll 4
    for (int s = 0; s < S; ++s) acc += a[s] * e[(size_t)s * H];
    x16[(size_t)b * 2 * H + c]     = tok16[(size_t)b * ldt + c];
    x16[(size_t)b * 2 * H + H + c] = (half_t)acc;
}

// ---------------------------------------------------------------------------
// Host orchestration (stream-ordered, graph-capture safe)
// ---------------------------------------------------------------------------
extern "C" void kernel_launch(void* const* d_in, const int* in_sizes, int n_in,
                              void* d_out, int out_size, void* d_ws, size_t ws_size,
                              hipStream_t stream)
{
    (void)in_sizes; (void)n_in; (void)out_size; (void)ws_size;

    const int*   tok_in   = (const int*)  d_in[0];
    const int*   tok_tgt  = (const int*)  d_in[1];
    const float* emb      = (const float*)d_in[2];
    const float* enc_W_ih = (const float*)d_in[3];
    const float* enc_W_hh = (const float*)d_in[4];
    const float* enc_b_ih = (const float*)d_in[5];
    const float* enc_b_hh = (const float*)d_in[6];
    const float* dec_W_ih = (const float*)d_in[7];
    const float* dec_W_hh = (const float*)d_in[8];
    const float* dec_b_ih = (const float*)d_in[9];
    const float* dec_b_hh = (const float*)d_in[10];
    const float* fc_W     = (const float*)d_in[11];
    const float* fc_b     = (const float*)d_in[12];
    float* out = (float*)d_out;

    char* ws = (char*)d_ws;
    size_t off = 0;
    auto alloc = [&](size_t bytes) -> void* {
        void* p = ws + off;
        off = (off + bytes + 255) & ~(size_t)255;
        return p;
    };

    half_t* fcW16    = (half_t*)alloc((size_t)V  * H * 2);
    half_t* encWih16 = (half_t*)alloc((size_t)H3 * H * 2);
    half_t* encWhh16 = (half_t*)alloc((size_t)H3 * H * 2);
    half_t* decWih16 = (half_t*)alloc((size_t)H3 * 2 * H * 2);
    half_t* decWhh16 = (half_t*)alloc((size_t)H3 * H * 2);
    half_t* srcEmb16 = (half_t*)alloc((size_t)Bsz * S * H * 2);
    float*  decEmbF  = (float*) alloc((size_t)Bsz * (T - 1) * H * 4);
    half_t* decEmb16 = (half_t*)alloc((size_t)Bsz * (T - 1) * H * 2);
    float*  giAll    = (float*) alloc((size_t)Bsz * S * H3 * 4);
    float*  encOut   = (float*) alloc((size_t)Bsz * S * H * 4);
    float*  hbuf     = (float*) alloc((size_t)Bsz * H * 4);
    half_t* h16      = (half_t*)alloc((size_t)Bsz * H * 2);
    float*  gi       = (float*) alloc((size_t)Bsz * H3 * 4);
    float*  gh       = (float*) alloc((size_t)Bsz * H3 * 4);
    float*  scores   = (float*) alloc((size_t)Bsz * S * 4);
    half_t* x16      = (half_t*)alloc((size_t)Bsz * 2 * H * 2);

    auto cvt = [&](const float* src, half_t* dst, int n) {
        f32_to_f16_kernel<<<(n + 255) / 256, 256, 0, stream>>>(src, dst, n);
    };
    auto gemm = [&](const half_t* A, int lda, const half_t* Wm, int ldw,
                    const float* bias, float* C, long long ldc, long long coff,
                    int Mtiles, int N, int K) {
        int Ntiles  = N / 16;
        int ngroups = (Ntiles + TNW - 1) / TNW;
        dim3 grid((ngroups + 3) / 4, Mtiles);
        gemm_wmma_f16<<<grid, 128, 0, stream>>>(A, lda, Wm, ldw, bias,
                                                C, ldc, coff, K, Ntiles);
    };

    // --- one-time per call: weight conversion, gathers, init -----------------
    cvt(fc_W,     fcW16,    V  * H);
    cvt(enc_W_ih, encWih16, H3 * H);
    cvt(enc_W_hh, encWhh16, H3 * H);
    cvt(dec_W_ih, decWih16, H3 * 2 * H);
    cvt(dec_W_hh, decWhh16, H3 * H);

    gather_src_kernel<<<(Bsz * S * H + 255) / 256, 256, 0, stream>>>(tok_in, emb, srcEmb16);
    gather_dec_kernel<<<(Bsz * (T - 1) * H + 255) / 256, 256, 0, stream>>>(tok_tgt, emb, decEmbF, decEmb16);
    init_h_kernel<<<(Bsz * H + 255) / 256, 256, 0, stream>>>(hbuf, h16);
    zero_row0_kernel<<<(Bsz * V + 255) / 256, 256, 0, stream>>>(out);

    // gi_all = src_emb @ enc_W_ih^T + b_ih   (M = B*S = 2048)
    gemm(srcEmb16, H, encWih16, H, enc_b_ih, giAll, H3, 0, (Bsz * S) / 16, H3, H);

    // --- encoder scan --------------------------------------------------------
    for (int s = 0; s < S; ++s) {
        gemm(h16, H, encWhh16, H, enc_b_hh, gh, H3, 0, 1, H3, H);
        gru_elem_kernel<<<(Bsz * H + 255) / 256, 256, 0, stream>>>(
            giAll + (size_t)s * H3, S * H3, gh, hbuf, h16,
            encOut + (size_t)s * H, S * H);
    }

    // --- decoder scan --------------------------------------------------------
    for (int t = 0; t < T - 1; ++t) {
        attn_scores_kernel<<<(Bsz * S) / 8, 256, 0, stream>>>(
            encOut, decEmbF + (size_t)t * H, (T - 1) * H, scores);
        softmax128_kernel<<<Bsz, 128, 0, stream>>>(scores);
        attn_ctx_concat_kernel<<<(Bsz * H + 255) / 256, 256, 0, stream>>>(
            scores, encOut, decEmb16 + (size_t)t * H, (T - 1) * H, x16);

        gemm(x16, 2 * H, decWih16, 2 * H, dec_b_ih, gi, H3, 0, 1, H3, 2 * H);
        gemm(h16, H,     decWhh16, H,     dec_b_hh, gh, H3, 0, 1, H3, H);
        gru_elem_kernel<<<(Bsz * H + 255) / 256, 256, 0, stream>>>(
            gi, H3, gh, hbuf, h16, nullptr, 0);

        // logits -> out[:, t+1, :]
        gemm(h16, H, fcW16, H, fc_b, out, (long long)T * V,
             (long long)(t + 1) * V, 1, V, H);
    }
}
